// MossAudioTokenizerVectorQuantize_46849503264983
// MI455X (gfx1250) — matrix operand order repaired and checked by reading, not compiled, not run on hardware
//
#include <hip/hip_runtime.h>
#include <math.h>

#define B_  8
#define D_  1024
#define T_  2048
#define CD_ 256
#define CS_ 8192

typedef __attribute__((ext_vector_type(16))) __bf16 v16bf;
typedef __attribute__((ext_vector_type(8)))  float  v8f;

union Frag32 { uint4 q[2]; v16bf v; };

__device__ __forceinline__ unsigned short f32_to_bf16(float f) {
  unsigned int u = __float_as_uint(f);
  u += 0x7FFFu + ((u >> 16) & 1u);          // round-to-nearest-even
  return (unsigned short)(u >> 16);
}

// A-fragment, 16x32 bf16, row-major source with leading dim `ld` (elements).
// Contiguous K layout used here (matches B): lanes 0-15 row=lane, K=0..15; lanes 16-31 row=lane-16, K=16..31.
__device__ __forceinline__ v16bf load_frag_a(const unsigned short* p, int ld, int lane) {
  const int r = lane & 15, hi = lane >> 4;
  const unsigned short* row = p + (size_t)r * ld + hi * 16;
  Frag32 f;
  f.q[0] = *(const uint4*)(row);        // K = hi*16 + 0..7
  f.q[1] = *(const uint4*)(row + 8);    // K = hi*16 + 8..15
  return f.v;
}

// B-fragment, 32x16 bf16, column `colp` contiguous in K.
// Contiguous K layout used here (matches A): lanes 0-15 col=lane K=0..15; lanes 16-31 col=lane-16 K=16..31.
__device__ __forceinline__ v16bf load_frag_b(const unsigned short* colp, int lane) {
  const int hi = lane >> 4;
  const unsigned short* p = colp + hi * 16;
  Frag32 f;
  f.q[0] = *(const uint4*)(p);
  f.q[1] = *(const uint4*)(p + 8);
  return f.v;
}

__device__ __forceinline__ v8f wmma_bf16(v16bf a, v16bf b, v8f c) {
  return __builtin_amdgcn_wmma_f32_16x16x32_bf16(false, a, false, b, (short)0, c, false, false);
}

// CDNA5 async global->LDS copy (ASYNCcnt-tracked, no VGPR data path).
// dst_lds: LDS byte offset (low 32 bits of the generic LDS-aperture address).
__device__ __forceinline__ void async_copy_b128(unsigned dst_lds, const void* src) {
  asm volatile("global_load_async_to_lds_b128 %0, %1, off"
               :: "v"(dst_lds), "v"(src)
               : "memory");
}
__device__ __forceinline__ void wait_asynccnt0() {
  asm volatile("s_wait_asynccnt 0" ::: "memory");
}

// ---------------- prep kernels ----------------

// z (B,D,T) f32 -> z_t (B,T,D) bf16 (makes WMMA B-fragments contiguous)
__global__ void __launch_bounds__(256) k_zt(const float* __restrict__ z,
                                            unsigned short* __restrict__ zt) {
  size_t i = (size_t)blockIdx.x * 256 + threadIdx.x;
  int t = (int)(i % T_);
  int d = (int)((i / T_) % D_);
  int b = (int)(i / ((size_t)T_ * D_));
  zt[((size_t)b * T_ + t) * D_ + d] = f32_to_bf16(z[i]);
}

// weight-norm: w = g * v / ||v||_row, output bf16. One block per row.
__global__ void __launch_bounds__(256) k_wnorm(const float* __restrict__ v,
                                               const float* __restrict__ g,
                                               unsigned short* __restrict__ w, int K) {
  __shared__ float red[256];
  const int row = blockIdx.x;
  const float* vr = v + (size_t)row * K;
  float s = 0.f;
  for (int k = threadIdx.x; k < K; k += 256) { float x = vr[k]; s += x * x; }
  red[threadIdx.x] = s;
  __syncthreads();
  for (int o = 128; o > 0; o >>= 1) {
    if (threadIdx.x < o) red[threadIdx.x] += red[threadIdx.x + o];
    __syncthreads();
  }
  const float scale = g[row] / sqrtf(red[0]);
  for (int k = threadIdx.x; k < K; k += 256)
    w[(size_t)row * K + k] = f32_to_bf16(vr[k] * scale);
}

// codebook row -> bf16 copy + ||c||^2. One block (256 thr) per code row (CD_=256).
__global__ void __launch_bounds__(256) k_cb(const float* __restrict__ cb,
                                            unsigned short* __restrict__ cbb,
                                            float* __restrict__ cn) {
  __shared__ float red[256];
  const int row = blockIdx.x;
  float x = cb[(size_t)row * CD_ + threadIdx.x];
  cbb[(size_t)row * CD_ + threadIdx.x] = f32_to_bf16(x);
  red[threadIdx.x] = x * x;
  __syncthreads();
  for (int o = 128; o > 0; o >>= 1) {
    if (threadIdx.x < o) red[threadIdx.x] += red[threadIdx.x + o];
    __syncthreads();
  }
  if (threadIdx.x == 0) cn[row] = red[0];
}

// ---------------- WMMA GEMM (in_proj / out_proj) ----------------
// Out[b][row][t] = sum_k A[row][k] * Bcol(t)[k] + bias[row]
// Bcol(t) = GATHER ? cbb + idx[b*T+t]*KTOT : zt + (b*T+t)*KTOT   (stride = KTOT both ways)
// Block: 256 thr = 8 waves (2x4); block tile 128x128; wave tile 64x32 (4x2 WMMA tiles).
template <int KTOT, bool GATHER, bool WRITE_ENC>
__global__ void __launch_bounds__(256)
gemm_bf16_kernel(const unsigned short* __restrict__ A,
                 const unsigned short* __restrict__ Bsrc,
                 const int* __restrict__ gidx,
                 const float* __restrict__ bias,
                 float* __restrict__ Out,
                 unsigned short* __restrict__ enc_out,
                 int M) {
  const int lane = threadIdx.x & 31;
  const int wave = threadIdx.x >> 5;
  const int b = blockIdx.z;
  const int mbase = blockIdx.x * 128 + (wave & 1) * 64;
  const int nbase = blockIdx.y * 128 + (wave >> 1) * 32;
  const int c = lane & 15, hi = lane >> 4;

  const unsigned short* colp[2];
#pragma unroll
  for (int nt = 0; nt < 2; ++nt) {
    const int t = nbase + nt * 16 + c;
    size_t col;
    if (GATHER) col = (size_t)gidx[b * T_ + t] * KTOT;
    else        col = ((size_t)b * T_ + t) * KTOT;
    colp[nt] = Bsrc + col;
  }

  v8f acc[4][2];
#pragma unroll
  for (int mt = 0; mt < 4; ++mt)
#pragma unroll
    for (int nt = 0; nt < 2; ++nt)
#pragma unroll
      for (int i = 0; i < 8; ++i) acc[mt][nt][i] = 0.f;

  for (int k0 = 0; k0 < KTOT; k0 += 32) {
    const v16bf bf0 = load_frag_b(colp[0] + k0, lane);
    const v16bf bf1 = load_frag_b(colp[1] + k0, lane);
#pragma unroll
    for (int mt = 0; mt < 4; ++mt) {
      const v16bf af = load_frag_a(A + (size_t)(mbase + mt * 16) * KTOT + k0, KTOT, lane);
      acc[mt][0] = wmma_bf16(af, bf0, acc[mt][0]);
      acc[mt][1] = wmma_bf16(af, bf1, acc[mt][1]);
    }
  }

#pragma unroll
  for (int mt = 0; mt < 4; ++mt) {
#pragma unroll
    for (int i = 0; i < 8; ++i) {
      const int row = mbase + mt * 16 + hi * 8 + i;
      const float bi = bias[row];
#pragma unroll
      for (int nt = 0; nt < 2; ++nt) {
        const int t = nbase + nt * 16 + c;
        const float val = acc[mt][nt][i] + bi;
        Out[((size_t)b * M + row) * T_ + t] = val;
        if (WRITE_ENC)
          enc_out[((size_t)b * T_ + t) * CD_ + row] = f32_to_bf16(val);
      }
    }
  }
}

// ---------------- distance + argmin ----------------
// dist(n,c) = ||c||^2 - 2 * enc(n)·cb(c)   (row norm of enc is constant per row)
// Wave owns one 16-row m-tile; 512 codebook n-tiles streamed through LDS that is
// double-buffered and filled by async global->LDS copies (ASYNCcnt path).
// Two independent 4-deep WMMA accumulator chains for ILP.
__global__ void __launch_bounds__(256)
k_dist(const unsigned short* __restrict__ enc,
       const unsigned short* __restrict__ cbb,
       const float* __restrict__ cn,
       float* __restrict__ idx_f,
       int* __restrict__ idx_i) {
  __shared__ unsigned short tileB[2][16 * CD_];   // 2 x 8KB
  const int lane = threadIdx.x & 31;
  const int wave = threadIdx.x >> 5;
  const int c = lane & 15, hi = lane >> 4;
  const int mbase = (blockIdx.x * 8 + wave) * 16;

  // LDS byte offsets of the two buffers for this thread's two 16B chunks.
  const unsigned ldsBase0 = (unsigned)(size_t)(&tileB[0][0]);
  const unsigned ldsBase1 = (unsigned)(size_t)(&tileB[1][0]);
  const unsigned myOff = (unsigned)threadIdx.x * 16u;

  // Preload all A fragments (16 rows x 256 K) into VGPRs: 8 ksteps.
  v16bf afr[8];
#pragma unroll
  for (int ks = 0; ks < 8; ++ks)
    afr[ks] = load_frag_a(enc + (size_t)mbase * CD_ + ks * 32, CD_, lane);

  float best[8];
  int bidx[8];
#pragma unroll
  for (int i = 0; i < 8; ++i) { best[i] = 3.4e38f; bidx[i] = 0; }

  // Async-load tile 0 into buffer 0.
  {
    const char* src = (const char*)cbb;
    async_copy_b128(ldsBase0 + myOff, src + myOff);
    async_copy_b128(ldsBase0 + myOff + 4096u, src + myOff + 4096u);
  }
  wait_asynccnt0();
  __syncthreads();

  for (int tile = 0; tile < CS_ / 16; ++tile) {
    const int cur = tile & 1;
    if (tile + 1 < CS_ / 16) {   // async-prefetch next tile into other buffer
      const char* src = (const char*)(cbb + (size_t)(tile + 1) * 16 * CD_);
      const unsigned dstBase = (cur ? ldsBase0 : ldsBase1);
      async_copy_b128(dstBase + myOff, src + myOff);
      async_copy_b128(dstBase + myOff + 4096u, src + myOff + 4096u);
    }

    v8f acc0, acc1;
#pragma unroll
    for (int i = 0; i < 8; ++i) { acc0[i] = 0.f; acc1[i] = 0.f; }
#pragma unroll
    for (int ks = 0; ks < 4; ++ks) {
      const unsigned short* p0 = &tileB[cur][c * CD_ + (2 * ks) * 32 + hi * 16];
      const unsigned short* p1 = &tileB[cur][c * CD_ + (2 * ks + 1) * 32 + hi * 16];
      Frag32 f0, f1;
      f0.q[0] = *(const uint4*)(p0);
      f0.q[1] = *(const uint4*)(p0 + 8);
      f1.q[0] = *(const uint4*)(p1);
      f1.q[1] = *(const uint4*)(p1 + 8);
      acc0 = wmma_bf16(afr[2 * ks], f0.v, acc0);
      acc1 = wmma_bf16(afr[2 * ks + 1], f1.v, acc1);
    }

    const float cnv = cn[tile * 16 + c];
    const int code = tile * 16 + c;
#pragma unroll
    for (int i = 0; i < 8; ++i) {
      const float d = cnv - 2.0f * (acc0[i] + acc1[i]);
      if (d < best[i]) { best[i] = d; bidx[i] = code; }
    }

    wait_asynccnt0();   // prefetched tile landed in LDS
    __syncthreads();
  }

  // Reduce across the 16 lanes holding the columns of each row.
#pragma unroll
  for (int i = 0; i < 8; ++i) {
    float v = best[i];
    int id = bidx[i];
    for (int m = 8; m >= 1; m >>= 1) {
      const float ov = __shfl_xor(v, m, 32);
      const int oid = __shfl_xor(id, m, 32);
      if (ov < v || (ov == v && oid < id)) { v = ov; id = oid; }
    }
    if (c == 0) {
      const int n = mbase + hi * 8 + i;
      idx_f[n] = (float)id;
      idx_i[n] = id;
    }
  }
}

// ---------------- launch ----------------

extern "C" void kernel_launch(void* const* d_in, const int* in_sizes, int n_in,
                              void* d_out, int out_size, void* d_ws, size_t ws_size,
                              hipStream_t stream) {
  const float* z     = (const float*)d_in[0];
  const float* in_v  = (const float*)d_in[1];
  const float* in_g  = (const float*)d_in[2];
  const float* in_b  = (const float*)d_in[3];
  const float* out_v = (const float*)d_in[4];
  const float* out_g = (const float*)d_in[5];
  const float* out_b = (const float*)d_in[6];
  const float* cb    = (const float*)d_in[7];

  char* ws = (char*)d_ws;
  unsigned short* zt   = (unsigned short*)(ws + 0);          // 32 MB  z transposed bf16
  unsigned short* win  = (unsigned short*)(ws + 33554432);   // 512 KB w_in bf16
  unsigned short* wout = (unsigned short*)(ws + 34078720);   // 512 KB w_out bf16
  unsigned short* cbb  = (unsigned short*)(ws + 34603008);   // 4 MB   codebook bf16
  float*          cn   = (float*)(ws + 38797312);            // 32 KB  ||c||^2
  unsigned short* enc  = (unsigned short*)(ws + 38830080);   // 8 MB   enc bf16 (N x CD)
  int*            idxi = (int*)(ws + 47218688);              // 64 KB  argmin indices

  float* zq   = (float*)d_out;                         // (B, D, T)
  float* idxf = zq + (size_t)B_ * D_ * T_;             // (B, T) as float
  float* ze   = idxf + (size_t)B_ * T_;                // (B, CD, T)

  k_zt<<<(B_ * D_ * T_) / 256, 256, 0, stream>>>(z, zt);
  k_wnorm<<<CD_, 256, 0, stream>>>(in_v, in_g, win, D_);
  k_wnorm<<<D_, 256, 0, stream>>>(out_v, out_g, wout, CD_);
  k_cb<<<CS_, 256, 0, stream>>>(cb, cbb, cn);

  // in_proj: z_e = w_in @ z + in_b ; also emit bf16 enc (N x CD)
  gemm_bf16_kernel<D_, false, true>
      <<<dim3(CD_ / 128, T_ / 128, B_), 256, 0, stream>>>(win, zt, nullptr, in_b, ze, enc, CD_);

  // nearest codebook entry per encoding
  k_dist<<<(B_ * T_) / (16 * 8), 256, 0, stream>>>(enc, cbb, cn, idxf, idxi);

  // out_proj: z_q = w_out @ codebook[idx] + out_b
  gemm_bf16_kernel<CD_, true, false>
      <<<dim3(D_ / 128, T_ / 128, B_), 256, 0, stream>>>(wout, cbb, idxi, out_b, zq, nullptr, D_);
}
